// CarbonGNN_36447092474498
// MI455X (gfx1250) — matrix operand order repaired and checked by reading, not compiled, not run on hardware
//
#include <hip/hip_runtime.h>
#include <hip/hip_bf16.h>

// ---------------------------------------------------------------------------
// GNN pipeline: GCNConv -> ReLU -> GATv2Conv(H=4, concat=False mean) -> ReLU
//               -> Linear(64->1)
// N=50000, E=800000, IN=128, HID=64, H=4.
// Dense GEMMs use V_WMMA_F32_16X16X32_F16 (f16 in, f32 accumulate), with a
// 64x64 block tile: 4 waves x (16 rows, 4 accumulators) so each A fragment
// feeds 4 back-to-back WMMAs.
// ---------------------------------------------------------------------------

typedef __attribute__((ext_vector_type(16))) _Float16 v16h;
typedef __attribute__((ext_vector_type(8)))  float    v8f;

#define IN_DIM   128
#define HID_DIM  64
#define HEADS    4
#define FEAT256  256   // HEADS*HID
#define NEG_SLOPE 0.2f

// --- monotone float <-> uint key for atomicMax on floats --------------------
__device__ __forceinline__ unsigned fkey(float f) {
    unsigned u = __float_as_uint(f);
    return (u & 0x80000000u) ? ~u : (u | 0x80000000u);
}
__device__ __forceinline__ float funkey(unsigned k) {
    unsigned u = (k & 0x80000000u) ? (k & 0x7FFFFFFFu) : ~k;
    return __uint_as_float(u);
}

// ---------------------------------------------------------------------------
// WMMA GEMM: C[M,Nc] = A[M,K](f32) * B[K,Nc](f32) + bias
// K multiple of 32, Nc multiple of 64. Block = 128 threads = 4 waves.
// Block tile = 64 rows x 64 cols; wave = 16 rows x 64 cols (4 accumulators).
// Grid = (ceil(M/64), Nc/64). B^T staged once per block into LDS as f16.
// A loads are clamped (not predicated) so the hot loop has no exec-mask ops;
// only the stores are guarded against the M tail.
// ---------------------------------------------------------------------------
__global__ __launch_bounds__(128)
void gemm_wmma_f16(const float* __restrict__ A, const float* __restrict__ Bw,
                   const float* __restrict__ bias, float* __restrict__ C,
                   int M, int K, int Nc) {
    __shared__ __align__(32) _Float16 ldsB[64 * 128];   // up to K=128

    const int tid     = threadIdx.x;
    const int rowBlk  = blockIdx.x * 64;
    const int colBase = blockIdx.y * 64;

    // stage B^T (64 local cols x K) into LDS as f16
    for (int idx = tid; idx < 64 * K; idx += 128) {
        int n = idx / K;
        int k = idx - n * K;
        ldsB[idx] = (_Float16)Bw[(size_t)k * Nc + colBase + n];
    }
    __syncthreads();

    const int lane = tid & 31;
    const int wv   = tid >> 5;        // wave's 16-row slice within block
    const int m    = lane & 15;
    const int hi   = lane >> 4;       // lane half selects K grouping

    const int rowBase = rowBlk + wv * 16;
    const int row     = rowBase + m;
    const int rowLd   = row < M ? row : (M - 1);   // clamp: loads always valid
    const float* Arow = A + (size_t)rowLd * K;

    v8f acc0 = {0.f,0.f,0.f,0.f,0.f,0.f,0.f,0.f};
    v8f acc1 = acc0, acc2 = acc0, acc3 = acc0;

    const int nK = K >> 5;
    for (int kt = 0; kt < nK; ++kt) {
        const int kb = kt * 32;
        // A fragment (16x32 f16): lanes 0-15: K 0..15; lanes 16-31: K 16..31
        v16h a;
        const float* p0 = Arow + kb + hi * 16;       // K = kb+hi*16   .. +7
        const float* p1 = Arow + kb + 8 + hi * 16;   // K = kb+8+hi*16 .. +7
        #pragma unroll
        for (int t = 0; t < 8; ++t) {
            a[t]     = (_Float16)p0[t];
            a[8 + t] = (_Float16)p1[t];
        }
        // B fragments (32x16 f16): lane = N col, halves = contiguous K run
        const _Float16* bbase = ldsB + kb + hi * 16;
        const v16h b0 = *(const v16h*)(bbase + (size_t)(m)      * K);
        acc0 = __builtin_amdgcn_wmma_f32_16x16x32_f16(false, a, false, b0,
                                                      (short)0, acc0, false, false);
        const v16h b1 = *(const v16h*)(bbase + (size_t)(16 + m) * K);
        acc1 = __builtin_amdgcn_wmma_f32_16x16x32_f16(false, a, false, b1,
                                                      (short)0, acc1, false, false);
        const v16h b2 = *(const v16h*)(bbase + (size_t)(32 + m) * K);
        acc2 = __builtin_amdgcn_wmma_f32_16x16x32_f16(false, a, false, b2,
                                                      (short)0, acc2, false, false);
        const v16h b3 = *(const v16h*)(bbase + (size_t)(48 + m) * K);
        acc3 = __builtin_amdgcn_wmma_f32_16x16x32_f16(false, a, false, b3,
                                                      (short)0, acc3, false, false);
    }

    // epilogue: C/D layout: lanes 0-15 hold M=r, lanes 16-31 hold M=8+r; N=m
    const float bv0 = bias ? bias[colBase + m]      : 0.f;
    const float bv1 = bias ? bias[colBase + 16 + m] : 0.f;
    const float bv2 = bias ? bias[colBase + 32 + m] : 0.f;
    const float bv3 = bias ? bias[colBase + 48 + m] : 0.f;
    #pragma unroll
    for (int r = 0; r < 8; ++r) {
        const int rr = rowBase + hi * 8 + r;
        if (rr < M) {
            float* crow = C + (size_t)rr * Nc + colBase + m;
            crow[0]  = acc0[r] + bv0;
            crow[16] = acc1[r] + bv1;
            crow[32] = acc2[r] + bv2;
            crow[48] = acc3[r] + bv3;
        }
    }
}

// ---------------------------------------------------------------------------
// Streaming kernels
// ---------------------------------------------------------------------------
__global__ void fill_f32(float* p, float v, long n) {
    long t = (long)blockIdx.x * blockDim.x + threadIdx.x;
    if (t < n) p[t] = v;
}

__global__ void deg_edges(const int* __restrict__ dst, float* deg, int E) {
    int t = blockIdx.x * blockDim.x + threadIdx.x;
    if (t < E) atomicAdd(&deg[dst[t]], 1.0f);
}

__global__ void rsqrt_k(float* d, int n) {
    int t = blockIdx.x * blockDim.x + threadIdx.x;
    if (t < n) d[t] = rsqrtf(d[t]);   // deg >= 1 always (self loop)
}

// h_acc[i,c] = dinv[i]^2 * xw[i,c]  (self-loop contribution)
__global__ void gcn_self(const float* __restrict__ xw, const float* __restrict__ dinv,
                         float* __restrict__ hacc, int n64) {
    int t = blockIdx.x * blockDim.x + threadIdx.x;
    if (t < n64) {
        int i = t >> 6;
        float w = dinv[i];
        hacc[t] = w * w * xw[t];
    }
}

// edge scatter: hacc[dst] += dinv[s]*dinv[d] * xw[src]  (float4 per thread)
__global__ void gcn_edges(const float* __restrict__ xw, const float* __restrict__ dinv,
                          const int* __restrict__ src, const int* __restrict__ dst,
                          float* __restrict__ hacc, int E) {
    int t = blockIdx.x * blockDim.x + threadIdx.x;
    if (t >= E * 16) return;
    int e  = t >> 4;
    int c4 = (t & 15) * 4;
    int s = src[e], d = dst[e];
    float w = dinv[s] * dinv[d];
    float4 v = *(const float4*)(xw + (size_t)s * HID_DIM + c4);
    float* hb = hacc + (size_t)d * HID_DIM + c4;
    atomicAdd(hb + 0, w * v.x);
    atomicAdd(hb + 1, w * v.y);
    atomicAdd(hb + 2, w * v.z);
    atomicAdd(hb + 3, w * v.w);
}

__global__ void relu_bias(float* __restrict__ h, const float* __restrict__ b, int n64) {
    int t = blockIdx.x * blockDim.x + threadIdx.x;
    if (t < n64) h[t] = fmaxf(h[t] + b[t & 63], 0.f);
}

// one wave per edge: logits[e,h] = sum_c att[h,c]*leaky(xl[s,h,c]+xr[d,h,c])
// + atomicMax segment-max per (dst, head)
__global__ void gat_logits(const float* __restrict__ xl, const float* __restrict__ xr,
                           const float* __restrict__ att,
                           const int* __restrict__ src, const int* __restrict__ dst,
                           float* __restrict__ logits, unsigned* __restrict__ mmax,
                           int E, int Etot) {
    int wave = (blockIdx.x * blockDim.x + threadIdx.x) >> 5;
    int lane = threadIdx.x & 31;
    if (wave >= Etot) return;
    int s, d;
    if (wave < E) { s = src[wave]; d = dst[wave]; } else { s = d = wave - E; }
    const float* pl = xl + (size_t)s * FEAT256;
    const float* pr = xr + (size_t)d * FEAT256;
    float sum[HEADS] = {0.f, 0.f, 0.f, 0.f};
    #pragma unroll
    for (int j = 0; j < 8; ++j) {
        int ch = lane + 32 * j;               // ch = head*64 + c
        float v = pl[ch] + pr[ch];
        v = v > 0.f ? v : NEG_SLOPE * v;
        sum[j >> 1] += att[ch] * v;
    }
    #pragma unroll
    for (int off = 16; off; off >>= 1) {
        sum[0] += __shfl_xor(sum[0], off);
        sum[1] += __shfl_xor(sum[1], off);
        sum[2] += __shfl_xor(sum[2], off);
        sum[3] += __shfl_xor(sum[3], off);
    }
    if (lane == 0) {
        float4* lp = (float4*)(logits + (size_t)wave * HEADS);
        *lp = make_float4(sum[0], sum[1], sum[2], sum[3]);
        unsigned* mp = mmax + (size_t)d * HEADS;
        atomicMax(mp + 0, fkey(sum[0]));
        atomicMax(mp + 1, fkey(sum[1]));
        atomicMax(mp + 2, fkey(sum[2]));
        atomicMax(mp + 3, fkey(sum[3]));
    }
}

// p = exp(logit - m[d]); denom[d,h] += p
__global__ void gat_expsum(float* __restrict__ logits, const unsigned* __restrict__ mmax,
                           float* __restrict__ denom, const int* __restrict__ dst,
                           int E, int Etot) {
    int t = blockIdx.x * blockDim.x + threadIdx.x;
    if (t >= Etot * HEADS) return;
    int e  = t >> 2;
    int hh = t & 3;
    int d  = (e < E) ? dst[e] : (e - E);
    float m = funkey(mmax[(size_t)d * HEADS + hh]);
    float p = expf(logits[t] - m);
    logits[t] = p;
    atomicAdd(&denom[(size_t)d * HEADS + hh], p);
}

// gacc[d, ch] += (p/denom) * xl[s, ch]   (float4 per thread)
__global__ void gat_scatter(const float* __restrict__ xl, const float* __restrict__ p,
                            const float* __restrict__ denom,
                            const int* __restrict__ src, const int* __restrict__ dst,
                            float* __restrict__ gacc, int E, int Etot) {
    int t = blockIdx.x * blockDim.x + threadIdx.x;
    if (t >= Etot * 64) return;
    int e   = t >> 6;
    int q   = t & 63;
    int ch4 = q * 4;
    int hh  = ch4 >> 6;
    int s, d;
    if (e < E) { s = src[e]; d = dst[e]; } else { s = d = e - E; }
    float alpha = p[(size_t)e * HEADS + hh] / denom[(size_t)d * HEADS + hh];
    float4 v = *(const float4*)(xl + (size_t)s * FEAT256 + ch4);
    float* g = gacc + (size_t)d * FEAT256 + ch4;
    atomicAdd(g + 0, alpha * v.x);
    atomicAdd(g + 1, alpha * v.y);
    atomicAdd(g + 2, alpha * v.z);
    atomicAdd(g + 3, alpha * v.w);
}

// h2[i,c] = relu(mean_h gacc[i,h,c] + b_gat[c])
__global__ void gat_mean_relu(const float* __restrict__ gacc, const float* __restrict__ b,
                              float* __restrict__ h2, int n64) {
    int t = blockIdx.x * blockDim.x + threadIdx.x;
    if (t >= n64) return;
    int i = t >> 6, c = t & 63;
    const float* g = gacc + (size_t)i * FEAT256;
    float v = 0.25f * (g[c] + g[64 + c] + g[128 + c] + g[192 + c]) + b[c];
    h2[t] = fmaxf(v, 0.f);
}

// out[i] = h2[i,:] . W_lin + b_lin   (one wave per node)
__global__ void head_out(const float* __restrict__ h2, const float* __restrict__ Wl,
                         const float* __restrict__ bl, float* __restrict__ out, int n) {
    int wave = (blockIdx.x * blockDim.x + threadIdx.x) >> 5;
    int lane = threadIdx.x & 31;
    if (wave >= n) return;
    const float* p = h2 + (size_t)wave * HID_DIM;
    float a = p[lane] * Wl[lane] + p[lane + 32] * Wl[lane + 32];
    #pragma unroll
    for (int off = 16; off; off >>= 1) a += __shfl_xor(a, off);
    if (lane == 0) out[wave] = a + bl[0];
}

// ---------------------------------------------------------------------------
extern "C" void kernel_launch(void* const* d_in, const int* in_sizes, int n_in,
                              void* d_out, int out_size, void* d_ws, size_t ws_size,
                              hipStream_t stream) {
    const float* x      = (const float*)d_in[0];
    const int*   eidx   = (const int*)d_in[1];
    // d_in[2] = edge_attr (unused by reference)
    const float* W_gcn  = (const float*)d_in[3];
    const float* b_gcn  = (const float*)d_in[4];
    const float* W_l    = (const float*)d_in[5];
    const float* b_l    = (const float*)d_in[6];
    const float* W_r    = (const float*)d_in[7];
    const float* b_r    = (const float*)d_in[8];
    const float* att    = (const float*)d_in[9];
    const float* b_gat  = (const float*)d_in[10];
    const float* W_lin  = (const float*)d_in[11];
    const float* b_lin  = (const float*)d_in[12];
    float*       out    = (float*)d_out;

    const int Nn   = in_sizes[0] / IN_DIM;
    const int E    = in_sizes[1] / 2;
    const int Etot = E + Nn;
    const int* src = eidx;
    const int* dst = eidx + E;

    // workspace layout (floats)
    float* ws = (float*)d_ws;
    size_t off = 0;
    float*    dinv   = ws + off; off += (size_t)Nn;            // deg -> dinv
    float*    xw     = ws + off; off += (size_t)Nn * HID_DIM;  // reused as h2
    float*    hacc   = ws + off; off += (size_t)Nn * HID_DIM;  // GCN out / GAT in
    float*    xl     = ws + off; off += (size_t)Nn * FEAT256;
    float*    xr     = ws + off; off += (size_t)Nn * FEAT256;
    float*    logits = ws + off; off += (size_t)Etot * HEADS;  // -> p
    unsigned* mmax   = (unsigned*)(ws + off); off += (size_t)Nn * HEADS;
    float*    denom  = ws + off; off += (size_t)Nn * HEADS;
    float*    gacc   = ws + off; off += (size_t)Nn * FEAT256;
    float*    h2     = xw;

    const int B = 256;
    #define NB(n) ((int)(((long)(n) + B - 1) / B))

    // --- init ---
    fill_f32<<<NB(Nn), B, 0, stream>>>(dinv, 1.0f, Nn);                     // deg (self loop)
    fill_f32<<<NB((long)Nn * HEADS), B, 0, stream>>>((float*)mmax, 0.0f, (long)Nn * HEADS);
    fill_f32<<<NB((long)Nn * HEADS), B, 0, stream>>>(denom, 0.0f, (long)Nn * HEADS);
    fill_f32<<<NB((long)Nn * FEAT256), B, 0, stream>>>(gacc, 0.0f, (long)Nn * FEAT256);

    // --- GCN ---
    deg_edges<<<NB(E), B, 0, stream>>>(dst, dinv, E);
    rsqrt_k<<<NB(Nn), B, 0, stream>>>(dinv, Nn);
    gemm_wmma_f16<<<dim3((Nn + 63) / 64, 1), 128, 0, stream>>>(
        x, W_gcn, nullptr, xw, Nn, IN_DIM, HID_DIM);
    gcn_self<<<NB(Nn * HID_DIM), B, 0, stream>>>(xw, dinv, hacc, Nn * HID_DIM);
    gcn_edges<<<NB((long)E * 16), B, 0, stream>>>(xw, dinv, src, dst, hacc, E);
    relu_bias<<<NB(Nn * HID_DIM), B, 0, stream>>>(hacc, b_gcn, Nn * HID_DIM);

    // --- GATv2 transforms ---
    gemm_wmma_f16<<<dim3((Nn + 63) / 64, FEAT256 / 64), 128, 0, stream>>>(
        hacc, W_l, b_l, xl, Nn, HID_DIM, FEAT256);
    gemm_wmma_f16<<<dim3((Nn + 63) / 64, FEAT256 / 64), 128, 0, stream>>>(
        hacc, W_r, b_r, xr, Nn, HID_DIM, FEAT256);

    // --- GATv2 attention ---
    gat_logits<<<(Etot + 7) / 8, B, 0, stream>>>(xl, xr, att, src, dst,
                                                 logits, mmax, E, Etot);
    gat_expsum<<<NB((long)Etot * HEADS), B, 0, stream>>>(logits, mmax, denom,
                                                         dst, E, Etot);
    gat_scatter<<<NB((long)Etot * 64), B, 0, stream>>>(xl, logits, denom,
                                                       src, dst, gacc, E, Etot);
    gat_mean_relu<<<NB(Nn * HID_DIM), B, 0, stream>>>(gacc, b_gat, h2, Nn * HID_DIM);

    // --- output head ---
    head_out<<<(Nn + 7) / 8, B, 0, stream>>>(h2, W_lin, b_lin, out, Nn);

    #undef NB
    (void)n_in; (void)out_size; (void)ws_size;
}
